// SO3SingleVector3_23708219474700
// MI455X (gfx1250) — compile-verified, matrix operand order checked by reading, not run on hardware
//
#include <hip/hip_runtime.h>
#include <hip/hip_bf16.h>

typedef __attribute__((ext_vector_type(16))) _Float16 v16h;
typedef __attribute__((ext_vector_type(8)))  float    v8f;

namespace {
constexpr int kNumBases = 10;
constexpr int kHid      = 256;
constexpr int kTiles    = kHid / 16;   // 16 column tiles of W1
}

__device__ __forceinline__ float relu1(float v) {
    // single v_med3_f32: clamp-below-zero (avoids fmaxf's canonicalize+max pair)
    return __builtin_amdgcn_fmed3f(v, 0.0f, 3.0e38f);
}

// Reduce 8 per-lane rows over a 16-lane half in 8 shuffles.
// On return, lane L holds the half-sum for row r = L&7.
__device__ __forceinline__ float foldreduce8(const float v[8], int lane) {
    const int b0 = lane & 1, b1 = (lane >> 1) & 1, b2 = (lane >> 2) & 1;
    float k0 = b0 ? v[1] : v[0];  float s0 = b0 ? v[0] : v[1];
    float k1 = b0 ? v[3] : v[2];  float s1 = b0 ? v[2] : v[3];
    float k2 = b0 ? v[5] : v[4];  float s2 = b0 ? v[4] : v[5];
    float k3 = b0 ? v[7] : v[6];  float s3 = b0 ? v[6] : v[7];
    k0 += __shfl_xor(s0, 1, 32);
    k1 += __shfl_xor(s1, 1, 32);
    k2 += __shfl_xor(s2, 1, 32);
    k3 += __shfl_xor(s3, 1, 32);
    float m0 = b1 ? k1 : k0;      float t0 = b1 ? k0 : k1;
    float m1 = b1 ? k3 : k2;      float t1 = b1 ? k2 : k3;
    m0 += __shfl_xor(t0, 2, 32);
    m1 += __shfl_xor(t1, 2, 32);
    float f = b2 ? m1 : m0;       float g = b2 ? m0 : m1;
    f += __shfl_xor(g, 4, 32);
    f += __shfl_xor(f, 8, 32);    // fold the two 8-lane groups of this half
    return f;                     // r = 4*b2 + 2*b1 + b0 = lane&7
}

__global__ __launch_bounds__(256) void so3_wmma_kernel(
    const float* __restrict__ x,
    const float* __restrict__ W1,
    const float* __restrict__ W2,
    float* __restrict__ out,
    int n)
{
    // W1 staged as pre-swizzled f16 B-fragments: (tile, lane) -> 16 halves (32 B)
    __shared__ __align__(32) unsigned int sB[kTiles * 32 * 8];   // 16 KB
    __shared__ float sW2[kHid * 2];                              // 2 KB

    const int tid  = threadIdx.x;
    const int lane = tid & 31;

    // ---- stage W2 * sqrt(2)/16  (sqrt(2) h-scale, 1/16 = 1/sqrt(HID)) ----
    for (int i = tid; i < kHid * 2; i += blockDim.x)
        sW2[i] = W2[i] * (1.41421356237f / 16.0f);

    // ---- stage W1 into f16 B-fragment layout (K padded 10 -> 32) ----
    // 16-bit A/B layout: lanes 0-15: halves 0..7 -> K=0..7,  halves 8..15 -> K=16..23
    //                    lanes16-31: halves 0..7 -> K=8..15, halves 8..15 -> K=24..31
    for (int u = tid; u < kTiles * 32 * 8; u += blockDim.x) {
        const int pair = u & 7;
        const int L    = (u >> 3) & 31;
        const int t    = u >> 8;
        const int nCol = 16 * t + (L & 15);
        unsigned int packed = 0;
        #pragma unroll
        for (int hh = 0; hh < 2; ++hh) {
            const int i  = pair * 2 + hh;
            const int kk = (L < 16) ? ((i < 8) ? i : (8 + i))
                                    : ((i < 8) ? (8 + i) : (16 + i));
            const float v = (kk < kNumBases) ? W1[kk * kHid + nCol] : 0.0f;
            const _Float16 hf = (_Float16)v;
            packed |= ((unsigned int)__builtin_bit_cast(unsigned short, hf)) << (16 * hh);
        }
        sB[u] = packed;
    }
    __syncthreads();

    // Each wave handles 32 points = two 16-row WMMA M-tiles per iteration.
    const int numChunks  = (n + 31) >> 5;
    const int wavesTotal = gridDim.x * (blockDim.x >> 5);
    const int waveId     = blockIdx.x * (blockDim.x >> 5) + (tid >> 5);

    const int   m        = lane & 15;
    const float stepInv  = 11.0f / 3.0f;          // 1/step, step = 3/(NUM_BASES+1)
    const float SMOOTH_C = 8.4335730f;            // 1.14136 * e^2
    const float s15      = 3.87298334621f;
    const float s5       = 2.23606797750f;

    for (int chunk = waveId; chunk < numChunks; chunk += wavesTotal) {
        const int base = chunk * 32;

        // ---- per-point data for both M-tiles (set0: rows base+m, set1: base+16+m) ----
        float xr[2], yr[2], zr[2], sh4[2], sh5[2], sh6[2], sh7[2], sh8[2], nrm[2];
        #pragma unroll
        for (int s = 0; s < 2; ++s) {
            const int p  = base + 16 * s + m;
            const int pc = (p < n) ? p : (n - 1);      // clamp: EXEC stays all-ones
            xr[s] = x[3 * pc + 0];
            yr[s] = x[3 * pc + 1];
            zr[s] = x[3 * pc + 2];
            const float nr = sqrtf(xr[s]*xr[s] + yr[s]*yr[s] + zr[s]*zr[s]);
            nrm[s] = nr;
            const float rinv = 1.0f / fmaxf(nr, 1e-20f);
            const float xn = xr[s]*rinv, yn = yr[s]*rinv, zn = zr[s]*rinv;
            sh4[s] = s15 * xn * zn;
            sh5[s] = s15 * xn * yn;
            sh6[s] = s5  * (yn*yn - 0.5f*(xn*xn + zn*zn));
            sh7[s] = s15 * yn * zn;
            sh8[s] = 0.5f * s15 * (zn*zn - xn*xn);
        }

        // ---- radial embed -> two A fragments (sqrt(10) factors cancel with W1) ----
        v16h afrag[2];
        #pragma unroll
        for (int s = 0; s < 2; ++s) {
            const float u0 = nrm[s] * stepInv;
            #pragma unroll
            for (int i = 0; i < 8; ++i) {
                const int   kk = (lane < 16) ? i : (8 + i);
                const float td = u0 - (float)(kk + 1);
                const float d1 = td + 1.0f, d2 = 1.0f - td;
                const float s1 = (d1 > 0.0f) ? __expf(-1.0f / d1) : 0.0f;
                const float s2 = (d2 > 0.0f) ? __expf(-1.0f / d2) : 0.0f;
                const float e  = (kk < kNumBases) ? (SMOOTH_C * s1 * s2) : 0.0f;
                afrag[s][i] = (_Float16)e;
            }
            #pragma unroll
            for (int i = 8; i < 16; ++i) afrag[s][i] = (_Float16)0.0f;
        }

        float acc[2][2][8];   // [set][j][row]
        #pragma unroll
        for (int s = 0; s < 2; ++s)
            #pragma unroll
            for (int j = 0; j < 2; ++j)
                #pragma unroll
                for (int r = 0; r < 8; ++r) acc[s][j][r] = 0.0f;

        // fused GEMM2 epilogue for one finished tile (ReLU in C layout + 2-col GEMV)
        auto tileEpilogue = [&](int t, const v8f& c0, const v8f& c1) {
            const int nCol = 16 * t + m;
            const float w2a = sW2[2 * nCol + 0];
            const float w2b = sW2[2 * nCol + 1];
            #pragma unroll
            for (int r = 0; r < 8; ++r) {
                const float h0 = relu1(c0[r]);
                const float h1 = relu1(c1[r]);
                acc[0][0][r] = fmaf(h0, w2a, acc[0][0][r]);
                acc[0][1][r] = fmaf(h0, w2b, acc[0][1][r]);
                acc[1][0][r] = fmaf(h1, w2a, acc[1][0][r]);
                acc[1][1][r] = fmaf(h1, w2b, acc[1][1][r]);
            }
        };

        // ---- two-stage pipeline: B prefetched 1 tile ahead, epilogue delayed 1 tile
        //      so a full epilogue sits between each WMMA and its D-register read.
        v16h bnxt = *(const v16h*)&sB[lane * 8];                 // tile 0
        v16h bfut = *(const v16h*)&sB[(32 + lane) * 8];          // tile 1
        v8f c0p = {}, c1p = {};
        c0p = __builtin_amdgcn_wmma_f32_16x16x32_f16(
                  false, afrag[0], false, bnxt, (short)0, c0p, false, false);
        c1p = __builtin_amdgcn_wmma_f32_16x16x32_f16(
                  false, afrag[1], false, bnxt, (short)0, c1p, false, false);
        bnxt = bfut;

        #pragma unroll 5
        for (int t = 1; t < kTiles; ++t) {
            const int tn = (t + 1) & (kTiles - 1);
            bfut = *(const v16h*)&sB[(tn * 32 + lane) * 8];      // prefetch t+1

            v8f c0 = {}, c1 = {};
            c0 = __builtin_amdgcn_wmma_f32_16x16x32_f16(
                     false, afrag[0], false, bnxt, (short)0, c0, false, false);
            c1 = __builtin_amdgcn_wmma_f32_16x16x32_f16(
                     false, afrag[1], false, bnxt, (short)0, c1, false, false);

            tileEpilogue(t - 1, c0p, c1p);                       // consume tile t-1

            c0p = c0; c1p = c1; bnxt = bfut;
        }
        tileEpilogue(kTiles - 1, c0p, c1p);                      // drain

        // ---- folded reduction over the 16-lane halves: 8 shuffles per (set,j),
        //      lane L ends holding row r = L&7 (matches the gather below) ----
        float sel[2][2];
        #pragma unroll
        for (int s = 0; s < 2; ++s)
            #pragma unroll
            for (int j = 0; j < 2; ++j)
                sel[s][j] = foldreduce8(acc[s][j], lane);

        // gather: lane L stores point base+L.
        const int srcA = m + ((m >= 8) ? 16 : 0);      // source for set0 values
        const int srcB = (m < 8) ? m : (m + 8);        // source for set1 values
        const float g00 = __shfl(sel[0][0], srcA, 32);
        const float g01 = __shfl(sel[0][1], srcA, 32);
        const float g10 = __shfl(sel[1][0], srcB, 32);
        const float g11 = __shfl(sel[1][1], srcB, 32);

        const int  s  = lane >> 4;                     // which M-tile this lane stores
        const float w0  = (s == 0) ? g00 : g10;
        const float w1v = (s == 0) ? g01 : g11;
        const float xs = xr[s],  ys = yr[s],  zs = zr[s];
        const float h4 = sh4[s], h5 = sh5[s], h6 = sh6[s], h7 = sh7[s], h8 = sh8[s];

        const int p = base + lane;
        if (p < n) {
            // C2 entries (Frobenius-normalized): cA=sqrt(0.1), cB=sqrt(1.5)*2/(3*sqrt(5))
            const float cA = 0.31622776601f;
            const float cB = 0.36514837167f;
            const float cC = 0.18257418583f;
            const float o2x = cA*(h4*zs + h5*ys) - cC*h6*xs - cA*h8*xs;
            const float o2y = cA*(h5*xs + h7*zs) + cB*h6*ys;
            const float o2z = cA*(h4*xs + h7*ys) - cC*h6*zs + cA*h8*zs;
            const float PATH_ALPHA = 1.22474487139f;   // sqrt(1.5)
            const float INV_SQRT3  = 0.57735026919f;
            const float a0 = PATH_ALPHA * INV_SQRT3 * w0;
            const float a2 = PATH_ALPHA * w1v;
            out[3 * p + 0] = a0 * xs + a2 * o2x;
            out[3 * p + 1] = a0 * ys + a2 * o2y;
            out[3 * p + 2] = a0 * zs + a2 * o2z;
        }
    }
}

extern "C" void kernel_launch(void* const* d_in, const int* in_sizes, int n_in,
                              void* d_out, int out_size, void* d_ws, size_t ws_size,
                              hipStream_t stream) {
    (void)n_in; (void)d_ws; (void)ws_size; (void)out_size;
    const float* x  = (const float*)d_in[0];
    const float* W1 = (const float*)d_in[1];
    const float* W2 = (const float*)d_in[2];
    float* out = (float*)d_out;

    const int n      = in_sizes[0] / 3;
    const int chunks = (n + 31) / 32;            // 32 points per wave iteration
    const int wavesPerBlock = 8;                 // 256 threads = 8 wave32
    int blocks = (chunks + wavesPerBlock - 1) / wavesPerBlock;
    if (blocks > 1024) blocks = 1024;            // grid-stride over remaining chunks
    if (blocks < 1) blocks = 1;
    so3_wmma_kernel<<<blocks, 256, 0, stream>>>(x, W1, W2, out, n);
}
